// MoE_37641093382396
// MI455X (gfx1250) — compile-verified, matrix operand order checked by reading, not run on hardware
//
#include <hip/hip_runtime.h>

#define NTOK 8192
#define DIMD 1024
#define HIDD 4096
#define NEXP 8
#define CAPE 1281          // int(1.25*8192/8)+1
#define CAPR 1408          // padded to 11 * 128 token tiles
#define BKP  48            // padded LDS K-stride (keeps 16B alignment, spreads banks)

typedef __attribute__((ext_vector_type(16))) __bf16 v16bf;
typedef __attribute__((ext_vector_type(8)))  float  v8f;

struct U128 { unsigned x, y, z, w; };

__device__ __forceinline__ unsigned short f2bf(float f) {
  unsigned u = __float_as_uint(f);
  u += 0x7FFFu + ((u >> 16) & 1u);           // round-to-nearest-even
  return (unsigned short)(u >> 16);
}

// 16x32 bf16 WMMA fragment: per-lane two contiguous 16B chunks
// (row = lane&15, K half selected by lane>>4), per CDNA5 ISA 7.12.2.
__device__ __forceinline__ v16bf ld_frag(const unsigned short* p) {
  union { U128 q[2]; v16bf v; } u;
  u.q[0] = *(const U128*)p;
  u.q[1] = *(const U128*)(p + 16);
  return u.v;
}

// ---------------------------------------------------------------- zero
__global__ void moe_zero(float* __restrict__ out, size_t n, float* __restrict__ hdr) {
  size_t i = blockIdx.x * (size_t)blockDim.x + threadIdx.x;
  const size_t stride = (size_t)gridDim.x * blockDim.x;
  for (; i < n; i += stride) out[i] = 0.0f;
  if (blockIdx.x == 0 && threadIdx.x < 256) hdr[threadIdx.x] = 0.0f;
}

// ---------------------------------------------------------------- fp32 -> bf16 weight pre-conversion
// One-shot pass so the GEMM hot loop stages pure bf16 (half the L2 traffic,
// zero conversion VALU per tile re-read).
__global__ void __launch_bounds__(256) moe_wconv(
    const float* __restrict__ src, unsigned short* __restrict__ dst, size_t n)
{
  size_t i = (blockIdx.x * (size_t)blockDim.x + threadIdx.x) * 8;
  const size_t stride = (size_t)gridDim.x * blockDim.x * 8;
  for (; i < n; i += stride) {
    const float4 f0 = *(const float4*)(src + i);
    const float4 f1 = *(const float4*)(src + i + 4);
    U128 q;
    q.x = (unsigned)f2bf(f0.x) | ((unsigned)f2bf(f0.y) << 16);
    q.y = (unsigned)f2bf(f0.z) | ((unsigned)f2bf(f0.w) << 16);
    q.z = (unsigned)f2bf(f1.x) | ((unsigned)f2bf(f1.y) << 16);
    q.w = (unsigned)f2bf(f1.z) | ((unsigned)f2bf(f1.w) << 16);
    *(U128*)(dst + i) = q;
  }
}

// ---------------------------------------------------------------- gating
// One wave per token: 8 dot products over D=1024, softmax, top-2.
__global__ void __launch_bounds__(256) moe_gate(
    const float* __restrict__ x, const float* __restrict__ gw,
    int* __restrict__ top_idx, float* __restrict__ top_w,
    float* __restrict__ me_sums)
{
  const int wid = threadIdx.x >> 5;
  const int lane = threadIdx.x & 31;
  const int t = blockIdx.x * 8 + wid;

  const float4* xr = (const float4*)(x + (size_t)t * DIMD);
  float acc[NEXP];
#pragma unroll
  for (int e = 0; e < NEXP; ++e) acc[e] = 0.0f;

#pragma unroll
  for (int it = 0; it < DIMD / 128; ++it) {
    const float4 xv = xr[it * 32 + lane];
#pragma unroll
    for (int e = 0; e < NEXP; ++e) {
      const float4 gv = ((const float4*)(gw + (size_t)e * DIMD))[it * 32 + lane];
      acc[e] += xv.x * gv.x + xv.y * gv.y + xv.z * gv.z + xv.w * gv.w;
    }
  }
#pragma unroll
  for (int e = 0; e < NEXP; ++e)
#pragma unroll
    for (int off = 16; off > 0; off >>= 1)
      acc[e] += __shfl_xor(acc[e], off, 32);

  if (lane == 0) {
    float mx = acc[0];
#pragma unroll
    for (int e = 1; e < NEXP; ++e) mx = fmaxf(mx, acc[e]);
    float p[NEXP], sum = 0.0f;
#pragma unroll
    for (int e = 0; e < NEXP; ++e) { p[e] = __expf(acc[e] - mx); sum += p[e]; }
    const float inv = 1.0f / sum;
#pragma unroll
    for (int e = 0; e < NEXP; ++e) p[e] *= inv;

    int e1 = 0; float b1 = p[0];
#pragma unroll
    for (int e = 1; e < NEXP; ++e) if (p[e] > b1) { b1 = p[e]; e1 = e; }   // first-index ties
    int e2 = (e1 == 0) ? 1 : 0; float b2 = p[e2];
#pragma unroll
    for (int e = 0; e < NEXP; ++e)
      if (e != e1 && p[e] > b2) { b2 = p[e]; e2 = e; }

    top_idx[2 * t] = e1; top_idx[2 * t + 1] = e2;
    top_w[2 * t] = b1;  top_w[2 * t + 1] = b2;
#pragma unroll
    for (int e = 0; e < NEXP; ++e) atomicAdd(&me_sums[e], p[e]);
  }
}

// ---------------------------------------------------------------- scan / capacity assignment
// Single 1024-thread block; 8 tokens/thread; Hillis-Steele scan per expert
// reproduces the reference's token-index-order ranks. Also emits aux loss.
__global__ void __launch_bounds__(1024) moe_scan(
    const int* __restrict__ top_idx, const float* __restrict__ top_w,
    int* __restrict__ idx_buf, float* __restrict__ w_buf,
    int* __restrict__ cnt_full, int* __restrict__ cnt_cap,
    const float* __restrict__ me_sums, float* __restrict__ out)
{
  const int tid = threadIdx.x;
  __shared__ int sbuf[1024];
  __shared__ int totals[NEXP];

  int cnt[NEXP];
#pragma unroll
  for (int e = 0; e < NEXP; ++e) cnt[e] = 0;
  const int t0 = tid * (NTOK / 1024);
  for (int k = 0; k < NTOK / 1024; ++k) {
    const int t = t0 + k;
    cnt[top_idx[2 * t]]++;
    cnt[top_idx[2 * t + 1]]++;
  }

  int pre[NEXP];
  for (int e = 0; e < NEXP; ++e) {
    sbuf[tid] = cnt[e];
    __syncthreads();
    for (int off = 1; off < 1024; off <<= 1) {
      int v = (tid >= off) ? sbuf[tid - off] : 0;
      __syncthreads();
      sbuf[tid] += v;
      __syncthreads();
    }
    pre[e] = sbuf[tid] - cnt[e];                 // exclusive prefix
    if (tid == 1023) totals[e] = sbuf[1023];
    __syncthreads();
  }

  for (int k = 0; k < NTOK / 1024; ++k) {
    const int t = t0 + k;
#pragma unroll
    for (int s = 0; s < 2; ++s) {
      const int e = top_idx[2 * t + s];
      const int r = pre[e]++;
      if (r < CAPE) {
        idx_buf[e * CAPR + r] = t;
        w_buf[e * CAPR + r] = top_w[2 * t + s];
      }
    }
  }

  if (tid < NEXP) {
    cnt_full[tid] = totals[tid];
    cnt_cap[tid] = totals[tid] < CAPE ? totals[tid] : CAPE;
  }
  if (tid == 0) {
    float aux = 0.0f;
    for (int e = 0; e < NEXP; ++e)
      aux += (me_sums[e] / (float)NTOK) * ((float)totals[e] / (float)NTOK);
    out[(size_t)NTOK * DIMD] = aux * (float)NEXP * 0.01f;   // tuple slot 2
  }
}

// ---------------------------------------------------------------- gather -> bf16 x_e
__global__ void __launch_bounds__(256) moe_gather(
    const float* __restrict__ x, const int* __restrict__ idx_buf,
    const int* __restrict__ cnt_cap, unsigned short* __restrict__ xe)
{
  const int row = blockIdx.x;                 // e * CAPR + slot
  const int e = row / CAPR;
  const int slot = row - e * CAPR;
  unsigned* dst = (unsigned*)(xe + (size_t)row * DIMD) + threadIdx.x * 2;
  if (slot < cnt_cap[e]) {
    const int tok = idx_buf[row];
    const float4 v = *((const float4*)(x + (size_t)tok * DIMD) + threadIdx.x);
    dst[0] = (unsigned)f2bf(v.x) | ((unsigned)f2bf(v.y) << 16);
    dst[1] = (unsigned)f2bf(v.z) | ((unsigned)f2bf(v.w) << 16);
  } else {
    dst[0] = 0u; dst[1] = 0u;                 // zero padding rows -> silu(0)=0
  }
}

// ---------------------------------------------------------------- WMMA GEMM
// MODE 0: h = silu(x_e @ w1b^T)  (A bf16 [CAPR,DIM], B bf16 [H,DIM], out bf16)
// MODE 1: out += w * (h @ w2b^T) (A bf16 [CAPR,HID], B bf16 [D,HID], atomic scatter)
// Staging uses gfx1250 async global->LDS copies (ASYNCcnt-tracked), double-buffered.
template<int MODE>
__global__ void __launch_bounds__(256) moe_gemm(
    const unsigned short* __restrict__ Abase,
    const unsigned short* __restrict__ Bbase,
    const int Kdim,
    const int* __restrict__ cnt_cap,
    unsigned short* __restrict__ Hout,
    float* __restrict__ out,
    const int* __restrict__ idx_buf,
    const float* __restrict__ w_buf)
{
  const int e = blockIdx.z;
  const int cnt = cnt_cap[e];
  const int m0 = blockIdx.x * 128;
  if (m0 >= cnt) return;                       // expert capacity early-exit
  const int n0 = blockIdx.y * 128;
  const int Nr = (MODE == 0) ? HIDD : DIMD;

  __shared__ unsigned short Asm[2][128][BKP];
  __shared__ unsigned short Bsm[2][128][BKP];

  const unsigned short* Aexp = Abase + ((size_t)e * CAPR + m0) * (size_t)Kdim;
  const unsigned short* Bexp = Bbase + ((size_t)e * Nr + n0) * (size_t)Kdim;

  const int tid = threadIdx.x;
  const int sr = tid >> 1;                     // staging row 0..127
  const int sc = (tid & 1) * 16;               // staging col (elements) 0/16

  // Async copy of one 128x32 bf16 tile pair into LDS buffer `b`.
  // Each lane issues 4x global_load_async_to_lds_b128; INST_OFFSET is added to
  // both the global and LDS addresses, so offset:16 covers the second chunk.
  auto stage = [&](int b, int k0) {
    const unsigned aoff = (unsigned)(size_t)&Asm[b][sr][sc];   // flat[31:0] == LDS offset
    const unsigned boff = (unsigned)(size_t)&Bsm[b][sr][sc];
    const unsigned long long ga =
        (unsigned long long)(size_t)(Aexp + (size_t)sr * Kdim + k0 + sc);
    const unsigned long long gb =
        (unsigned long long)(size_t)(Bexp + (size_t)sr * Kdim + k0 + sc);
    asm volatile(
        "global_load_async_to_lds_b128 %0, %2, off\n\t"
        "global_load_async_to_lds_b128 %0, %2, off offset:16\n\t"
        "global_load_async_to_lds_b128 %1, %3, off\n\t"
        "global_load_async_to_lds_b128 %1, %3, off offset:16"
        :: "v"(aoff), "v"(boff), "v"(ga), "v"(gb)
        : "memory");
  };

  const int wid  = tid >> 5;
  const int wm   = (wid >> 1) * 32;            // wave M offset (4 rows of waves)
  const int wn   = (wid & 1) * 64;             // wave N offset (2 cols of waves)
  const int lane = tid & 31;
  const int lm   = lane & 15;
  const int lhi  = lane >> 4;

  v8f acc[2][4];
#pragma unroll
  for (int i = 0; i < 2; ++i)
#pragma unroll
    for (int j = 0; j < 4; ++j)
#pragma unroll
      for (int k = 0; k < 8; ++k) acc[i][j][k] = 0.0f;

  stage(0, 0);
  asm volatile("s_wait_asynccnt 0x0" ::: "memory");
  __syncthreads();

  const int nK = Kdim / 32;
  for (int kt = 0; kt < nK; ++kt) {
    const int cur = kt & 1;
    if (kt + 1 < nK) stage(cur ^ 1, (kt + 1) * 32);   // async prefetch next K block

    v16bf af[2], bfr[4];
#pragma unroll
    for (int i = 0; i < 2; ++i)
      af[i] = ld_frag(&Asm[cur][wm + i * 16 + lm][lhi * 8]);
#pragma unroll
    for (int j = 0; j < 4; ++j)
      bfr[j] = ld_frag(&Bsm[cur][wn + j * 16 + lm][lhi * 8]);

#pragma unroll
    for (int i = 0; i < 2; ++i)
#pragma unroll
      for (int j = 0; j < 4; ++j)
        acc[i][j] = __builtin_amdgcn_wmma_f32_16x16x32_bf16(
            false, af[i], false, bfr[j], (short)0, acc[i][j], false, false);

    asm volatile("s_wait_asynccnt 0x0" ::: "memory");  // own async group done
    __syncthreads();                                   // all waves' groups done
  }

  if (MODE == 0) {
    unsigned short* H = Hout + ((size_t)e * CAPR + m0) * (size_t)HIDD + n0;
#pragma unroll
    for (int i = 0; i < 2; ++i)
#pragma unroll
      for (int r = 0; r < 8; ++r) {
        const int ml = wm + i * 16 + lhi * 8 + r;      // C layout: VGPR r, lane half
#pragma unroll
        for (int j = 0; j < 4; ++j) {
          const float v = acc[i][j][r];
          const float s = v / (1.0f + __expf(-v));     // silu
          H[(size_t)ml * HIDD + wn + j * 16 + lm] = f2bf(s);
        }
      }
  } else {
    const int base = e * CAPR;
#pragma unroll
    for (int i = 0; i < 2; ++i)
#pragma unroll
      for (int r = 0; r < 8; ++r) {
        const int slot = m0 + wm + i * 16 + lhi * 8 + r;
        if (slot < cnt) {
          const int tok = idx_buf[base + slot];
          const float wv = w_buf[base + slot];
          float* orow = out + (size_t)tok * DIMD + n0 + wn + lm;
#pragma unroll
          for (int j = 0; j < 4; ++j)
            unsafeAtomicAdd(orow + j * 16, acc[i][j][r] * wv);
        }
      }
  }
}

// ---------------------------------------------------------------- host
extern "C" void kernel_launch(void* const* d_in, const int* in_sizes, int n_in,
                              void* d_out, int out_size, void* d_ws, size_t ws_size,
                              hipStream_t stream) {
  (void)in_sizes; (void)n_in; (void)out_size; (void)ws_size;
  const float* x  = (const float*)d_in[0];
  const float* gw = (const float*)d_in[1];
  const float* w1 = (const float*)d_in[2];
  const float* w2 = (const float*)d_in[3];
  float* out = (float*)d_out;
  char* ws = (char*)d_ws;

  float* me       = (float*)(ws + 0);                                   //  8 f32
  int*   cnt_full = (int*)(ws + 256);                                   //  8 i32
  int*   cnt_cap  = (int*)(ws + 512);                                   //  8 i32
  int*   top_idx  = (int*)(ws + 1024);                                  //  N*2 i32
  float* top_w    = (float*)(ws + 1024 + (size_t)NTOK * 2 * 4);         //  N*2 f32
  int*   idx_buf  = (int*)(ws + 1024 + (size_t)NTOK * 4 * 4);           //  E*CAPR i32
  float* w_buf    = (float*)((char*)idx_buf + (size_t)NEXP * CAPR * 4); //  E*CAPR f32
  unsigned short* xe  = (unsigned short*)((char*)w_buf + (size_t)NEXP * CAPR * 4); // bf16 x_e
  unsigned short* hb  = xe + (size_t)NEXP * CAPR * DIMD;                // bf16 hidden (88 MiB, L2-resident)
  unsigned short* w1b = hb + (size_t)NEXP * CAPR * HIDD;                // bf16 w1
  unsigned short* w2b = w1b + (size_t)NEXP * HIDD * DIMD;               // bf16 w2

  const size_t wElems = (size_t)NEXP * HIDD * DIMD;

  moe_zero<<<8192, 256, 0, stream>>>(out, (size_t)NTOK * DIMD + 1, me);
  moe_wconv<<<8192, 256, 0, stream>>>(w1, w1b, wElems);
  moe_wconv<<<8192, 256, 0, stream>>>(w2, w2b, wElems);
  moe_gate<<<NTOK / 8, 256, 0, stream>>>(x, gw, top_idx, top_w, me);
  moe_scan<<<1, 1024, 0, stream>>>(top_idx, top_w, idx_buf, w_buf,
                                   cnt_full, cnt_cap, me, out);
  moe_gather<<<NEXP * CAPR, 256, 0, stream>>>(x, idx_buf, cnt_cap, xe);
  moe_gemm<0><<<dim3(CAPR / 128, HIDD / 128, NEXP), 256, 0, stream>>>(
      xe, w1b, DIMD, cnt_cap, hb, nullptr, nullptr, nullptr);
  moe_gemm<1><<<dim3(CAPR / 128, DIMD / 128, NEXP), 256, 0, stream>>>(
      hb, w2b, HIDD, cnt_cap, nullptr, out, idx_buf, w_buf);
}
